// MambaLayer2D_25220047962473
// MI455X (gfx1250) — matrix operand order repaired and checked by reading, not compiled, not run on hardware
//
#include <hip/hip_runtime.h>
#include <hip/hip_bf16.h>
#include <math.h>

#define NB    4
#define CDIM  128
#define LSEQ  4096
#define DIN   256
#define E2    512
#define NST   16
#define RRANK 8
#define NDBL  40   // RRANK + 2*NST
#define NPAD  48   // NDBL padded to multiple of 16 for WMMA
#define NSEG  32   // scan segments
#define SLEN  128  // steps per segment (NSEG*SLEN == LSEQ)

typedef __attribute__((ext_vector_type(16))) __bf16 v16bf;
typedef __attribute__((ext_vector_type(8)))  __bf16 v8bf;
typedef __attribute__((ext_vector_type(8)))  float  v8f;

static __device__ __forceinline__ v16bf cat16(v8bf lo, v8bf hi) {
  return __builtin_shufflevector(lo, hi, 0,1,2,3,4,5,6,7,8,9,10,11,12,13,14,15);
}

// ---------------- LayerNorm over C, write bf16 activations ----------------
__global__ void k_layernorm(const float* __restrict__ x,
                            const float* __restrict__ nw,
                            const float* __restrict__ nb,
                            __bf16* __restrict__ xnb) {
  int bl = blockIdx.x;          // over B*L
  int b  = bl >> 12;            // L = 4096
  int c  = threadIdx.x;         // 0..127
  int l  = bl & (LSEQ - 1);
  float v = x[((size_t)(b * CDIM + c)) * LSEQ + l];
  float s = v, s2 = v * v;
  #pragma unroll
  for (int off = 16; off >= 1; off >>= 1) {
    s  += __shfl_xor(s,  off, 32);
    s2 += __shfl_xor(s2, off, 32);
  }
  __shared__ float sh[8];
  int w = threadIdx.x >> 5;
  if ((threadIdx.x & 31) == 0) { sh[w] = s; sh[4 + w] = s2; }
  __syncthreads();
  float ts = 0.f, ts2 = 0.f;
  #pragma unroll
  for (int i = 0; i < 4; ++i) { ts += sh[i]; ts2 += sh[4 + i]; }
  float mu  = ts * (1.0f / CDIM);
  float var = ts2 * (1.0f / CDIM) - mu * mu;
  float inv = rsqrtf(var + 1e-5f);
  float xn  = (v - mu) * inv * nw[c] + nb[c];
  xnb[(size_t)bl * CDIM + c] = (__bf16)xn;
}

// ---------------- cast GEMM weights fp32 -> bf16 (x_proj padded to 48 rows) ---------
__global__ void k_cast_w(const float* __restrict__ wi, const float* __restrict__ wo,
                         const float* __restrict__ xw,
                         __bf16* __restrict__ wib, __bf16* __restrict__ wob,
                         __bf16* __restrict__ xwb) {
  int i = blockIdx.x * blockDim.x + threadIdx.x;
  if (i < E2 * CDIM)  wib[i] = (__bf16)wi[i];
  if (i < CDIM * DIN) wob[i] = (__bf16)wo[i];
  if (i < NPAD * DIN) xwb[i] = (i < NDBL * DIN) ? (__bf16)xw[i] : (__bf16)0.0f;
}

// ---------------- WMMA GEMM: out[b, l, n] = sum_k A[b,l,k] * W[n,k] ----------------
// A: (B*L, KDIM) bf16 row-major; W: (N, KDIM) bf16 row-major.
// MODE 0 (in_proj):  n<DIN -> out0 (xin), else -> out1 (z), layout (B,L,DIN)
// MODE 1 (out_proj): write transposed into (B, CDIM, L)
// MODE 2 (x_proj):   store n<NDBL into dbl (B,L,NDBL)
template<int KDIM, int MT, int NT, int MODE>
__global__ void __launch_bounds__(32)
k_gemm_wmma(const __bf16* __restrict__ A,
            const __bf16* __restrict__ Wb,
            float* __restrict__ out0,
            float* __restrict__ out1) {
  constexpr int KC = KDIM / 32;
  int lane = threadIdx.x & 31;
  int half = lane >> 4;         // 0: lanes 0-15, 1: lanes 16-31
  int mrow = lane & 15;
  int m0 = blockIdx.x * (16 * MT);
  int n0 = blockIdx.y * (16 * NT);
  int b  = blockIdx.z;

  // B fragments (32x16, K x N): lane holds column n, K-range half*16..+15
  v16bf bfrag[NT][KC];
  #pragma unroll
  for (int nt = 0; nt < NT; ++nt) {
    const __bf16* wrow = Wb + (size_t)(n0 + nt * 16 + mrow) * KDIM + half * 16;
    #pragma unroll
    for (int kc = 0; kc < KC; ++kc) {
      v8bf b0 = *(const v8bf*)(wrow + kc * 32);
      v8bf b1 = *(const v8bf*)(wrow + kc * 32 + 8);
      bfrag[nt][kc] = cat16(b0, b1);
    }
  }

  const __bf16* abase = A + ((size_t)b * LSEQ + m0 + mrow) * KDIM + half * 16;
  for (int mt = 0; mt < MT; ++mt) {
    const __bf16* arow = abase + (size_t)mt * 16 * KDIM;
    if (mt + 1 < MT) __builtin_prefetch(arow + (size_t)16 * KDIM, 0, 1);
    v8f acc[NT] = {};
    #pragma unroll
    for (int kc = 0; kc < KC; ++kc) {
      // A fragment (16x32): elems 0-15 = K kc*32+half*16..+15
      v8bf a0 = *(const v8bf*)(arow + kc * 32);
      v8bf a1 = *(const v8bf*)(arow + kc * 32 + 8);
      v16bf afrag = cat16(a0, a1);
      #pragma unroll
      for (int nt = 0; nt < NT; ++nt)
        acc[nt] = __builtin_amdgcn_wmma_f32_16x16x32_bf16(
            false, afrag, false, bfrag[nt][kc], (short)0, acc[nt], false, false);
    }
    // D layout: VGPR r -> M = r + half*8 (rows), N = mrow (col)
    int lbase = m0 + mt * 16 + half * 8;
    #pragma unroll
    for (int nt = 0; nt < NT; ++nt) {
      int n = n0 + nt * 16 + mrow;
      if constexpr (MODE == 0) {
        float* dst = (n < DIN) ? out0 : out1;
        int e = n & (DIN - 1);
        #pragma unroll
        for (int r = 0; r < 8; ++r)
          dst[((size_t)b * LSEQ + lbase + r) * DIN + e] = acc[nt][r];
      } else if constexpr (MODE == 1) {
        #pragma unroll
        for (int r = 0; r < 8; ++r)
          out0[((size_t)(b * CDIM + n)) * LSEQ + lbase + r] = acc[nt][r];
      } else {
        if (n < NDBL) {
          #pragma unroll
          for (int r = 0; r < 8; ++r)
            out0[((size_t)b * LSEQ + lbase + r) * NDBL + n] = acc[nt][r];
        }
      }
    }
  }
  (void)out1;
}

// ---------------- depthwise causal conv (d_conv=4) + SiLU (fp32 + bf16 copies) ------
__global__ void k_conv_silu(const float* __restrict__ xin,
                            const float* __restrict__ cw,   // (DIN,4)
                            const float* __restrict__ cb,
                            float* __restrict__ xc,
                            __bf16* __restrict__ xcb) {
  size_t i = (size_t)blockIdx.x * blockDim.x + threadIdx.x;  // over B*L*DIN
  if (i >= (size_t)NB * LSEQ * DIN) return;
  int d = (int)(i & (DIN - 1));
  size_t bl = i >> 8;
  int l = (int)(bl & (LSEQ - 1));
  int b = (int)(bl >> 12);
  float acc = cb[d];
  #pragma unroll
  for (int k = 0; k < 4; ++k) {
    int ls = l - 3 + k;
    float xv = (ls >= 0) ? xin[((size_t)b * LSEQ + ls) * DIN + d] : 0.0f;
    acc = fmaf(xv, cw[d * 4 + k], acc);
  }
  float s = acc / (1.0f + __expf(-acc));   // SiLU
  xc[i]  = s;
  xcb[i] = (__bf16)s;
}

// ---------------- delta = softplus(dt @ dt_proj_w^T + b) ----------------
__global__ void k_delta(const float* __restrict__ dbl,
                        const float* __restrict__ dtw,  // (DIN, 8)
                        const float* __restrict__ dtb,
                        float* __restrict__ delta) {
  size_t i = (size_t)blockIdx.x * blockDim.x + threadIdx.x;  // over B*L*DIN
  if (i >= (size_t)NB * LSEQ * DIN) return;
  int d = (int)(i & (DIN - 1));
  size_t bl = i >> 8;
  const float* r = dbl + bl * NDBL;
  float acc = dtb[d];
  #pragma unroll
  for (int j = 0; j < RRANK; ++j) acc = fmaf(r[j], dtw[d * RRANK + j], acc);
  delta[i] = (acc > 20.f) ? acc : log1pf(__expf(acc));
}

// ============ Segmented parallel scan (h <- dA*h + dBu is linear in h) =============
// Pass A: per (b, d-pair, segment) wave: local transfer pair P = prod(dA), Q = h|h0=0
__global__ void __launch_bounds__(32)
k_scan_part(const float* __restrict__ delta,
            const float* __restrict__ xc,
            const float* __restrict__ dbl,
            const float* __restrict__ alog,
            float* __restrict__ Pb, float* __restrict__ Qb) {
  int lane = threadIdx.x;
  int half = lane >> 4;
  int n    = lane & 15;
  int s  = blockIdx.x;               // segment
  int dp = blockIdx.y;               // d pair
  int b  = blockIdx.z;
  int d  = dp * 2 + half;
  float a = -__expf(alog[d * NST + n]);
  float h = 0.0f, P = 1.0f;
  int l0 = s * SLEN;
  size_t base_ld = ((size_t)b * LSEQ) * DIN + d;
  size_t base_db = ((size_t)b * LSEQ) * NDBL;
  for (int t = 0; t < SLEN; ++t) {
    int l = l0 + t;
    float dl  = delta[base_ld + (size_t)l * DIN];
    float xcv = xc[base_ld + (size_t)l * DIN];
    float Bn  = dbl[base_db + (size_t)l * NDBL + RRANK + n];
    float dA  = __expf(dl * a);
    h = fmaf(dA, h, dl * Bn * xcv);
    P *= dA;
  }
  size_t ci = (((size_t)b * DIN + d) * NST + n) * NSEG + s;
  Pb[ci] = P;
  Qb[ci] = h;
}

// Pass B: per chain (b,d,n): 32-step sequential combine -> h_start per segment
__global__ void k_scan_combine(const float* __restrict__ Pb,
                               const float* __restrict__ Qb,
                               float* __restrict__ Hs) {
  int i = blockIdx.x * blockDim.x + threadIdx.x;   // over B*DIN*NST = 16384
  if (i >= NB * DIN * NST) return;
  size_t base = (size_t)i * NSEG;
  float h = 0.0f;
  for (int s = 0; s < NSEG; ++s) {
    Hs[base + s] = h;
    h = fmaf(Pb[base + s], h, Qb[base + s]);
  }
}

// Pass C: per (b, d-pair, segment) wave: re-scan from true h_start, emit y
__global__ void __launch_bounds__(32)
k_scan_y(const float* __restrict__ delta,
         const float* __restrict__ xc,
         const float* __restrict__ dbl,
         const float* __restrict__ alog,
         const float* __restrict__ Dv,
         const float* __restrict__ Hs,
         float* __restrict__ y) {
  int lane = threadIdx.x;
  int half = lane >> 4;
  int n    = lane & 15;
  int s  = blockIdx.x;
  int dp = blockIdx.y;
  int b  = blockIdx.z;
  int d  = dp * 2 + half;
  float a  = -__expf(alog[d * NST + n]);
  float Dd = Dv[d];
  float h  = Hs[(((size_t)b * DIN + d) * NST + n) * NSEG + s];
  int l0 = s * SLEN;
  size_t base_ld = ((size_t)b * LSEQ) * DIN + d;
  size_t base_db = ((size_t)b * LSEQ) * NDBL;
  for (int t = 0; t < SLEN; ++t) {
    int l = l0 + t;
    float dl  = delta[base_ld + (size_t)l * DIN];
    float xcv = xc[base_ld + (size_t)l * DIN];
    float Bn  = dbl[base_db + (size_t)l * NDBL + RRANK + n];
    float Cn  = dbl[base_db + (size_t)l * NDBL + RRANK + NST + n];
    float dA  = __expf(dl * a);
    h = fmaf(dA, h, dl * Bn * xcv);
    float contrib = h * Cn;
    // reduce over 16 states (stays within each 16-lane half of the wave)
    contrib += __shfl_xor(contrib, 1, 32);
    contrib += __shfl_xor(contrib, 2, 32);
    contrib += __shfl_xor(contrib, 4, 32);
    contrib += __shfl_xor(contrib, 8, 32);
    if (n == 0) y[base_ld + (size_t)l * DIN] = fmaf(xcv, Dd, contrib);
  }
}

// ---------------- gating: yg = y * silu(z), cast to bf16 ----------------
__global__ void k_gate(const float* __restrict__ y,
                       const float* __restrict__ z,
                       __bf16* __restrict__ ygb) {
  size_t i = (size_t)blockIdx.x * blockDim.x + threadIdx.x;  // over B*L*DIN
  if (i >= (size_t)NB * LSEQ * DIN) return;
  float zv = z[i];
  float g  = zv / (1.0f + __expf(-zv));
  ygb[i] = (__bf16)(y[i] * g);
}

extern "C" void kernel_launch(void* const* d_in, const int* in_sizes, int n_in,
                              void* d_out, int out_size, void* d_ws, size_t ws_size,
                              hipStream_t stream) {
  const float* x    = (const float*)d_in[0];
  const float* nw   = (const float*)d_in[1];
  const float* nb   = (const float*)d_in[2];
  const float* wi   = (const float*)d_in[3];
  const float* cw   = (const float*)d_in[4];
  const float* cb   = (const float*)d_in[5];
  const float* xw   = (const float*)d_in[6];
  const float* dtw  = (const float*)d_in[7];
  const float* dtb  = (const float*)d_in[8];
  const float* alog = (const float*)d_in[9];
  const float* Dv   = (const float*)d_in[10];
  const float* wo   = (const float*)d_in[11];

  char* ws = (char*)d_ws;
  size_t off = 0;
  auto take = [&](size_t bytes) -> char* {
    char* p = ws + off;
    off += (bytes + 255) & ~(size_t)255;
    return p;
  };
  __bf16* xnb   = (__bf16*)take((size_t)NB * LSEQ * CDIM * 2);
  __bf16* wib   = (__bf16*)take((size_t)E2 * CDIM * 2);
  __bf16* wob   = (__bf16*)take((size_t)CDIM * DIN * 2);
  __bf16* xwb   = (__bf16*)take((size_t)NPAD * DIN * 2);
  float*  xin   = (float*) take((size_t)NB * LSEQ * DIN * 4);
  float*  zbuf  = (float*) take((size_t)NB * LSEQ * DIN * 4);
  float*  xc    = (float*) take((size_t)NB * LSEQ * DIN * 4);
  __bf16* xcb   = (__bf16*)take((size_t)NB * LSEQ * DIN * 2);
  float*  dbl   = (float*) take((size_t)NB * LSEQ * NDBL * 4);
  float*  delta = (float*) take((size_t)NB * LSEQ * DIN * 4);
  float*  ybuf  = (float*) take((size_t)NB * LSEQ * DIN * 4);
  __bf16* ygb   = (__bf16*)take((size_t)NB * LSEQ * DIN * 2);
  float*  Pb    = (float*) take((size_t)NB * DIN * NST * NSEG * 4);
  float*  Qb    = (float*) take((size_t)NB * DIN * NST * NSEG * 4);
  float*  Hs    = (float*) take((size_t)NB * DIN * NST * NSEG * 4);

  // 1. LayerNorm -> bf16
  k_layernorm<<<NB * LSEQ, 128, 0, stream>>>(x, nw, nb, xnb);
  // 2. cast weights (in_proj, out_proj, x_proj padded)
  k_cast_w<<<(E2 * CDIM + 255) / 256, 256, 0, stream>>>(wi, wo, xw, wib, wob, xwb);
  // 3. in_proj GEMM (M=4096/b, N=512, K=128) -> xin, z   [2 N-tiles per wave]
  k_gemm_wmma<CDIM, 8, 2, 0><<<dim3(LSEQ / 128, E2 / 32, NB), 32, 0, stream>>>(xnb, wib, xin, zbuf);
  // 4. conv + silu (fp32 + bf16)
  {
    size_t nthr = (size_t)NB * LSEQ * DIN;
    k_conv_silu<<<(unsigned)((nthr + 255) / 256), 256, 0, stream>>>(xin, cw, cb, xc, xcb);
  }
  // 5. x_proj GEMM (M=4096/b, N=48 padded, K=256) -> dbl
  k_gemm_wmma<DIN, 8, 1, 2><<<dim3(LSEQ / 128, NPAD / 16, NB), 32, 0, stream>>>(xcb, xwb, dbl, nullptr);
  // 6. delta = softplus(dt_proj)
  {
    size_t nthr = (size_t)NB * LSEQ * DIN;
    k_delta<<<(unsigned)((nthr + 255) / 256), 256, 0, stream>>>(dbl, dtw, dtb, delta);
  }
  // 7. segmented scan: 16384 parallel waves (A), 32-step combine (B), 16384 waves (C)
  k_scan_part<<<dim3(NSEG, DIN / 2, NB), 32, 0, stream>>>(delta, xc, dbl, alog, Pb, Qb);
  k_scan_combine<<<(NB * DIN * NST + 255) / 256, 256, 0, stream>>>(Pb, Qb, Hs);
  k_scan_y<<<dim3(NSEG, DIN / 2, NB), 32, 0, stream>>>(delta, xc, dbl, alog, Dv, Hs, ybuf);
  // 8. gate + cast
  {
    size_t nthr = (size_t)NB * LSEQ * DIN;
    k_gate<<<(unsigned)((nthr + 255) / 256), 256, 0, stream>>>(ybuf, zbuf, ygb);
  }
  // 9. out_proj GEMM (M=4096/b, N=128, K=256) -> transposed (B, C, L) output
  k_gemm_wmma<DIN, 8, 2, 1><<<dim3(LSEQ / 128, CDIM / 32, NB), 32, 0, stream>>>(ygb, wob, (float*)d_out, nullptr);
  (void)in_sizes; (void)n_in; (void)out_size; (void)ws_size;
}
